// CausalSelfAttention_59545426592275
// MI455X (gfx1250) — matrix operand order repaired and checked by reading, not compiled, not run on hardware
//
#include <hip/hip_runtime.h>
#include <hip/hip_bf16.h>
#include <math.h>

#define TSEQ  2048
#define CEMB  2048
#define NHEAD 16
#define HDIM  128
#define BSZ   4

typedef unsigned short u16;
typedef unsigned int   u32;
typedef __attribute__((ext_vector_type(16))) __bf16 v16bf;
typedef __attribute__((ext_vector_type(8)))  float  v8f;

union FragBF { v16bf v; uint4 q[2]; };

// ---- gfx1250 Tensor Data Mover path (6-arg builtin: amdgpu-toolchain/clang-23)
#if defined(__has_builtin)
#if __has_builtin(__builtin_amdgcn_tensor_load_to_lds) && \
    __has_builtin(__builtin_amdgcn_s_wait_tensorcnt)
#define HAVE_TDM 1
#endif
#endif

#if defined(HAVE_TDM)
typedef __attribute__((ext_vector_type(4))) unsigned int v4u;
typedef __attribute__((ext_vector_type(8))) int v8i_t;
typedef __attribute__((ext_vector_type(4))) int v4i_t;

__device__ __forceinline__ u32 lds_off(const void* p) {
  // addrspace(3) pointers are 32-bit offsets from LDS base on amdgcn
  return (u32)(unsigned long long)(__attribute__((address_space(3))) const void*)p;
}

// TDM 2D tile load: `rows` x 32 bf16 tile from row-major [dim1][ld] at (r0,k0)
// into LDS offset ldsoff. pad: 16B after every 64B row -> 80B LDS row stride.
__device__ __forceinline__ void tdm_load_tile(const u16* base, int ld, int dim1,
                                              int r0, int k0, int rows,
                                              u32 ldsoff) {
  unsigned long long ga = (unsigned long long)(size_t)(base + (size_t)r0 * ld + k0);
  v4u g0;
  g0.x = 1u;                            // D# count = 1 (user descriptor)
  g0.y = ldsoff;                        // lds_addr
  g0.z = (u32)ga;                       // global_addr[31:0]
  g0.w = (u32)(ga >> 32) | (2u << 30);  // global_addr[56:32] | type=2 (image)
  v8i_t g1;
  g1[0] = (1 << 16)     // data_size = 2 bytes
        | (1 << 20)     // pad_enable
        | (3 << 22)     // pad_interval: 16 DWORDs (64B)
        | (3 << 25);    // pad_amount: 4 DWORDs (16B)
  g1[1] = (ld & 0xffff) << 16;                        // tensor_dim0[15:0]
  g1[2] = ((ld >> 16) & 0xffff) | ((dim1 & 0xffff) << 16);  // dim0 hi | dim1 lo
  g1[3] = ((dim1 >> 16) & 0xffff) | (32 << 16);       // dim1 hi | tile_dim0 = 32
  g1[4] = rows & 0xffff;                              // tile_dim1 = rows
  g1[5] = ld;                                         // tensor_dim0_stride lo32
  g1[6] = 0;
  g1[7] = 0;
  v4i_t z4 = {0, 0, 0, 0};
  v8i_t z8 = {0, 0, 0, 0, 0, 0, 0, 0};
  __builtin_amdgcn_tensor_load_to_lds(g0, g1, z4, z4, z8, 0);
}
#endif

__device__ __forceinline__ u16 f2bf(float f) {
  u32 u = __float_as_uint(f);
  u32 r = u + 0x7FFFu + ((u >> 16) & 1u);   // round-to-nearest-even
  return (u16)(r >> 16);
}

__device__ __forceinline__ v8f wmma_bf16(v16bf a, v16bf b, v8f c) {
  return __builtin_amdgcn_wmma_f32_16x16x32_bf16(false, a, false, b, (short)0, c,
                                                 false, false);
}

// A-fragment, 16x32 from row-major [rows][ld] bf16.
// lane l: M = l&15 ; K = b8 + e (e = 0..15, contiguous 32B); b8 = (l>=16)*16
__device__ __forceinline__ v16bf load_a_frag(const u16* base, int ld, int r0,
                                             int k0, int lane) {
  int b8 = (lane & 16) ? 16 : 0;
  const u16* p = base + (size_t)(r0 + (lane & 15)) * ld + k0 + b8;
  FragBF f;
  f.q[0] = *(const uint4*)(p);
  f.q[1] = *(const uint4*)(p + 8);
  return f.v;
}

// B-fragment, (N=16)x(K=32) from an [N][K] row-major (pre-transposed) source.
// lane l: N = l&15 ; K = (l>=16?16:0) + e  (e = 0..15, contiguous 32B)
__device__ __forceinline__ v16bf load_b_frag(const u16* base, int ld, int n0,
                                             int k0, int lane) {
  int kb = (lane & 16) ? 16 : 0;
  const u16* p = base + (size_t)(n0 + (lane & 15)) * ld + k0 + kb;
  FragBF f;
  f.q[0] = *(const uint4*)(p);
  f.q[1] = *(const uint4*)(p + 8);
  return f.v;
}

// ---------------- fp32 -> bf16 convert ----------------
__global__ void cvt_f32_bf16_kernel(const float* __restrict__ in,
                                    u16* __restrict__ out, int n) {
  int i = (blockIdx.x * blockDim.x + threadIdx.x) * 4;
  if (i < n) {
    float4 f = *(const float4*)(in + i);
    union { u16 s[4]; uint2 u; } r;
    r.s[0] = f2bf(f.x); r.s[1] = f2bf(f.y);
    r.s[2] = f2bf(f.z); r.s[3] = f2bf(f.w);
    *(uint2*)(out + i) = r.u;
  }
}

// ---------------- transpose [K][N] f32 -> [N][K] bf16 ----------------
__global__ void transpose_cvt_kernel(const float* __restrict__ in,
                                     u16* __restrict__ out, int K, int N) {
  __shared__ float tile[32][33];
  int kb = blockIdx.y * 32, nb = blockIdx.x * 32;
#pragma unroll
  for (int i = 0; i < 32; i += 8)
    tile[threadIdx.y + i][threadIdx.x] =
        in[(size_t)(kb + threadIdx.y + i) * N + nb + threadIdx.x];
  __syncthreads();
#pragma unroll
  for (int i = 0; i < 32; i += 8)
    out[(size_t)(nb + threadIdx.y + i) * K + kb + threadIdx.x] =
        f2bf(tile[threadIdx.x][threadIdx.y + i]);
}

// ---------------- RoPE cos/sin tables, [T][64] f32 ----------------
__global__ void rope_table_kernel(float* __restrict__ cosT,
                                  float* __restrict__ sinT) {
  int idx = blockIdx.x * blockDim.x + threadIdx.x;
  if (idx >= TSEQ * 64) return;
  int t = idx >> 6, f = idx & 63;
  float inv = expf(-((float)(2 * f) / 128.0f) * logf(10000.0f));
  float ang = (float)t * inv;
  cosT[idx] = cosf(ang);
  sinT[idx] = sinf(ang);
}

// ---------------- tiled bf16 GEMM: C[M][*] = A[M][K] * Bt[N][K]^T ----------------
// block = 256 thr = 8 waves. Block tile 256x128; wave tile 32 rows x 128 cols.
// Double-buffered LDS; TDM (tensor_load_to_lds) staging when available.
// MODE 0: proj  -> outF = acc + bias        (f32)
// MODE 1: qkv   -> rope(q,k) scatter to [B,H,T,D]; v scatter transposed [B,H,D,T]
template <int MODE>
__global__ __launch_bounds__(256) void gemm_kernel(
    const u16* __restrict__ A, const u16* __restrict__ Bt,
    const float* __restrict__ bias, float* __restrict__ outF,
    u16* __restrict__ qb, u16* __restrict__ kbuf, u16* __restrict__ vT,
    const float* __restrict__ cosT, const float* __restrict__ sinT, int Kdim,
    int Mrows, int Nrows) {
  __shared__ __align__(16) u16 lA[2][256 * 40];  // 80B row stride (bank-safe)
  __shared__ __align__(16) u16 lB[2][128 * 40];
  const int tid = threadIdx.x, lane = tid & 31, wave = tid >> 5;
  const int m0 = blockIdx.y * 256, n0 = blockIdx.x * 128;

  v8f acc[2][8];
  const v8f vzero = {0.f, 0.f, 0.f, 0.f, 0.f, 0.f, 0.f, 0.f};
#pragma unroll
  for (int mt = 0; mt < 2; ++mt)
#pragma unroll
    for (int nt = 0; nt < 8; ++nt) acc[mt][nt] = vzero;

  const int lr = tid >> 2;        // 0..63
  const int lc = (tid & 3) * 8;   // 0,8,16,24 (x8 bf16 = 16B)
  (void)lr; (void)lc;

  // stage K-step tile k0 into LDS buffer `buf`
  auto stage = [&](int buf, int k0) {
#if defined(HAVE_TDM)
    if (wave == 0) {  // one wave drives the DMA engine (EXEC-independent)
      tdm_load_tile(A, Kdim, Mrows, m0, k0, 256, lds_off(&lA[buf][0]));
      tdm_load_tile(Bt, Kdim, Nrows, n0, k0, 128, lds_off(&lB[buf][0]));
    }
#else
#pragma unroll
    for (int i = 0; i < 4; ++i) {
      int r = lr + i * 64;
      *(uint4*)(&lA[buf][0] + r * 40 + lc) =
          *(const uint4*)(A + (size_t)(m0 + r) * Kdim + k0 + lc);
    }
#pragma unroll
    for (int i = 0; i < 2; ++i) {
      int r = lr + i * 64;
      *(uint4*)(&lB[buf][0] + r * 40 + lc) =
          *(const uint4*)(Bt + (size_t)(n0 + r) * Kdim + k0 + lc);
    }
#endif
  };

  const int nk = Kdim / 32;
  stage(0, 0);
  for (int ks = 0; ks < nk; ++ks) {
    const int cur = ks & 1;
    const bool has_next = (ks + 1) < nk;
    if (has_next) stage(cur ^ 1, (ks + 1) * 32);  // prefetch next K-tile
#if defined(HAVE_TDM)
    if (wave == 0) {
      // TDM ops of one wave complete in order: <=2 outstanding => cur landed
      if (has_next) __builtin_amdgcn_s_wait_tensorcnt(2);
      else          __builtin_amdgcn_s_wait_tensorcnt(0);
    }
#endif
    __syncthreads();  // publish buffer `cur` to all waves
    const u16* sA = &lA[cur][0];
    const u16* sB = &lB[cur][0];
    v16bf af0 = load_a_frag(sA, 40, wave * 32, 0, lane);
    v16bf af1 = load_a_frag(sA, 40, wave * 32 + 16, 0, lane);
#pragma unroll
    for (int nt = 0; nt < 8; ++nt) {
      v16bf bfv = load_b_frag(sB, 40, nt * 16, 0, lane);
      acc[0][nt] = wmma_bf16(af0, bfv, acc[0][nt]);
      acc[1][nt] = wmma_bf16(af1, bfv, acc[1][nt]);
    }
    __syncthreads();  // done reading `cur` before it gets restaged
  }

  // epilogue: C layout -> VGPR i holds row (i + 8*(lane>=16)), col = lane&15
  const int rloc = (lane & 16) ? 8 : 0;
  const int cl = lane & 15;
  if (MODE == 0) {
#pragma unroll
    for (int mt = 0; mt < 2; ++mt)
#pragma unroll
      for (int i = 0; i < 8; ++i) {
        int gr = m0 + wave * 32 + mt * 16 + i + rloc;
#pragma unroll
        for (int nt = 0; nt < 8; ++nt) {
          int gc = n0 + nt * 16 + cl;
          outF[(size_t)gr * CEMB + gc] = acc[mt][nt][i] + bias[gc];
        }
      }
  } else {
    int sec = n0 / CEMB;            // 0=q 1=k 2=v (block = exactly one head)
    int h = (n0 % CEMB) / HDIM;
#pragma unroll
    for (int mt = 0; mt < 2; ++mt)
#pragma unroll
      for (int i = 0; i < 8; ++i) {
        int gr = m0 + wave * 32 + mt * 16 + i + rloc;
        int b = gr / TSEQ, t = gr % TSEQ;
        size_t hb = (size_t)(b * NHEAD + h);
        if (sec == 2) {
#pragma unroll
          for (int nt = 0; nt < 8; ++nt) {
            int d = nt * 16 + cl;
            float v = acc[mt][nt][i] + bias[n0 + d];
            vT[(hb * HDIM + d) * TSEQ + t] = f2bf(v);  // store v transposed
          }
        } else {
          u16* dst = (sec == 0 ? qb : kbuf) + (hb * TSEQ + t) * HDIM;
#pragma unroll
          for (int nt = 0; nt < 4; ++nt) {
            int f = nt * 16 + cl;  // pair (f, f+64) both in this wave tile
            float x1 = acc[mt][nt][i] + bias[n0 + f];
            float x2 = acc[mt][nt + 4][i] + bias[n0 + f + 64];
            float cv = cosT[t * 64 + f], sv = sinT[t * 64 + f];
            dst[f]      = f2bf(x1 * cv - x2 * sv);
            dst[f + 64] = f2bf(x2 * cv + x1 * sv);
          }
        }
      }
  }
}

// ---------------- flash attention: per wave 16 q-rows, k chunks of 32 ----------------
__global__ __launch_bounds__(256) void attn_kernel(const u16* __restrict__ qb,
                                                   const u16* __restrict__ kb,
                                                   const u16* __restrict__ vT,
                                                   u16* __restrict__ yb) {
  __shared__ __align__(16) u16 pS[8][16 * 40];  // per-wave prob tile 16x32 (+pad)
  const int lane = threadIdx.x & 31, wave = threadIdx.x >> 5;
  const int bh = blockIdx.y;
  const int qt0 = blockIdx.x * 128 + wave * 16;
  const u16* qh = qb + (size_t)bh * TSEQ * HDIM;
  const u16* kh = kb + (size_t)bh * TSEQ * HDIM;
  const u16* vh = vT + (size_t)bh * HDIM * TSEQ;
  u16* myP = &pS[wave][0];

  v16bf qf[4];
#pragma unroll
  for (int kk = 0; kk < 4; ++kk)
    qf[kk] = load_a_frag(qh, HDIM, qt0, kk * 32, lane);

  const v8f vzero = {0.f, 0.f, 0.f, 0.f, 0.f, 0.f, 0.f, 0.f};
  v8f y[8];
#pragma unroll
  for (int j = 0; j < 8; ++j) y[j] = vzero;
  float mrow[8], lrow[8];
#pragma unroll
  for (int i = 0; i < 8; ++i) { mrow[i] = -1e30f; lrow[i] = 0.f; }

  const float sc = 0.08838834764831845f;  // 1/sqrt(128)
  const int rbase = qt0 + ((lane & 16) ? 8 : 0);
  const int prow = (lane & 16) ? 8 : 0;
  const int cl = lane & 15;
  const int kend = qt0 + 16;

  for (int kc = 0; kc < kend; kc += 32) {
    v8f S0 = vzero, S1 = vzero;
#pragma unroll
    for (int kk = 0; kk < 4; ++kk) {
      v16bf k0f = load_b_frag(kh, HDIM, kc, kk * 32, lane);
      v16bf k1f = load_b_frag(kh, HDIM, kc + 16, kk * 32, lane);
      S0 = wmma_bf16(qf[kk], k0f, S0);
      S1 = wmma_bf16(qf[kk], k1f, S1);
    }
    const int col0 = kc + cl, col1 = col0 + 16;
#pragma unroll
    for (int i = 0; i < 8; ++i) {
      int r = rbase + i;
      float s0 = (col0 <= r) ? S0[i] * sc : -1e30f;
      float s1 = (col1 <= r) ? S1[i] * sc : -1e30f;
      float mx = fmaxf(s0, s1);
      mx = fmaxf(mx, __shfl_xor(mx, 1));
      mx = fmaxf(mx, __shfl_xor(mx, 2));
      mx = fmaxf(mx, __shfl_xor(mx, 4));
      mx = fmaxf(mx, __shfl_xor(mx, 8));
      float mnew = fmaxf(mrow[i], mx);
      float rescale = __expf(mrow[i] - mnew);
      float p0 = __expf(s0 - mnew);
      float p1 = __expf(s1 - mnew);
      float rs = p0 + p1;
      rs += __shfl_xor(rs, 1);
      rs += __shfl_xor(rs, 2);
      rs += __shfl_xor(rs, 4);
      rs += __shfl_xor(rs, 8);
      lrow[i] = lrow[i] * rescale + rs;
      mrow[i] = mnew;
#pragma unroll
      for (int j = 0; j < 8; ++j) y[j][i] *= rescale;
      myP[(prow + i) * 40 + cl]      = f2bf(p0);   // C-layout -> row-major LDS
      myP[(prow + i) * 40 + 16 + cl] = f2bf(p1);
    }
    asm volatile("s_wait_dscnt 0" ::: "memory");   // LDS stores visible to reads
    v16bf pf = load_a_frag(myP, 40, 0, 0, lane);   // re-read in A-fragment layout
#pragma unroll
    for (int j = 0; j < 8; ++j) {
      v16bf vf = load_b_frag(vh, TSEQ, j * 16, kc, lane);
      y[j] = wmma_bf16(pf, vf, y[j]);
    }
  }

  const int b = bh >> 4, h = bh & 15;
#pragma unroll
  for (int i = 0; i < 8; ++i) {
    float inv = 1.0f / lrow[i];
    int t = rbase + i;
    size_t rowoff = (size_t)(b * TSEQ + t) * CEMB + h * HDIM;
#pragma unroll
    for (int j = 0; j < 8; ++j)
      yb[rowoff + j * 16 + cl] = f2bf(y[j][i] * inv);
  }
}

extern "C" void kernel_launch(void* const* d_in, const int* in_sizes, int n_in,
                              void* d_out, int out_size, void* d_ws,
                              size_t ws_size, hipStream_t stream) {
  const float* x      = (const float*)d_in[0];
  const float* w_attn = (const float*)d_in[1];
  const float* b_attn = (const float*)d_in[2];
  const float* w_proj = (const float*)d_in[3];
  const float* b_proj = (const float*)d_in[4];
  float* out = (float*)d_out;
  (void)in_sizes; (void)n_in; (void)out_size; (void)ws_size;

  char* ws = (char*)d_ws;
  size_t off = 0;
  auto carve = [&](size_t bytes) -> char* {
    char* p = ws + off;
    off += (bytes + 255) & ~(size_t)255;
    return p;
  };
  const size_t M = (size_t)BSZ * TSEQ;  // 8192
  u16* xb     = (u16*)carve(M * CEMB * 2);
  u16* wqkvT  = (u16*)carve((size_t)3 * CEMB * CEMB * 2);
  u16* wprojT = (u16*)carve((size_t)CEMB * CEMB * 2);
  u16* qbuf   = (u16*)carve(M * CEMB * 2);
  u16* kbuf   = (u16*)carve(M * CEMB * 2);
  u16* vTbuf  = (u16*)carve(M * CEMB * 2);
  u16* ybuf   = (u16*)carve(M * CEMB * 2);
  float* cosT = (float*)carve((size_t)TSEQ * 64 * 4);
  float* sinT = (float*)carve((size_t)TSEQ * 64 * 4);

  int nx = (int)(M * CEMB);
  cvt_f32_bf16_kernel<<<nx / (256 * 4), 256, 0, stream>>>(x, xb, nx);
  transpose_cvt_kernel<<<dim3(3 * CEMB / 32, CEMB / 32), dim3(32, 8), 0, stream>>>(
      w_attn, wqkvT, CEMB, 3 * CEMB);
  transpose_cvt_kernel<<<dim3(CEMB / 32, CEMB / 32), dim3(32, 8), 0, stream>>>(
      w_proj, wprojT, CEMB, CEMB);
  rope_table_kernel<<<(TSEQ * 64) / 256, 256, 0, stream>>>(cosT, sinT);

  gemm_kernel<1><<<dim3(3 * CEMB / 128, (int)(M / 256)), 256, 0, stream>>>(
      xb, wqkvT, b_attn, nullptr, qbuf, kbuf, vTbuf, cosT, sinT, CEMB,
      (int)M, 3 * CEMB);

  attn_kernel<<<dim3(TSEQ / 128, BSZ * NHEAD), 256, 0, stream>>>(qbuf, kbuf,
                                                                 vTbuf, ybuf);

  gemm_kernel<0><<<dim3(CEMB / 128, (int)(M / 256)), 256, 0, stream>>>(
      ybuf, wprojT, b_proj, out, nullptr, nullptr, nullptr, nullptr, nullptr,
      CEMB, (int)M, CEMB);
}
